// LSTMModel_41412074668762
// MI455X (gfx1250) — matrix operand order repaired and checked by reading, not compiled, not run on hardware
//
#include <hip/hip_runtime.h>
#include <hip/hip_bf16.h>

// ---------------------------------------------------------------------------
// 2-layer bidirectional LSTM (B=2048, T=512, I=1, H=32) + Linear(64,2) head.
//
// MI455X (gfx1250, wave32) design:
//  * One wave owns a 16-row batch tile and runs the T=512 recurrence.
//    gates[16x128] = A[16x32/64] x B[KxN] via v_wmma_f32_16x16x32_f16,
//    weights converted to f16 once and held in VGPRs as B-fragments.
//  * h state round-trips through LDS each step (WMMA D layout -> A layout)
//    with two 16B ds loads per lane (contiguous K order k = 16*(lane/16)+s, as in B).
//  * Cell nonlinearities use gfx1250 V_TANH_F32 (branch-free), sigmoid as
//    0.5+0.5*tanh(x/2): ~5 trans ops/element, co-executing with WMMA.
//  * Layer-1 backward direction is algebraically ONE step per sequence
//    (bwd[b,len-1] == bwd_rev[b,0]).
//  * h1 stored fp16 (128 MB ws): ~0.27 GB HBM traffic (~12us @ 23.3 TB/s);
//    real limit is recurrence latency, hidden by 128-256 independent waves.
// ---------------------------------------------------------------------------

#define BATCH  2048
#define TSEQ   512
#define HID    32
#define TILE_M 16

typedef __attribute__((ext_vector_type(16))) _Float16 v16h;
typedef __attribute__((ext_vector_type(8)))  float    v8f;

union F16Frag {
    v16h     v;
    _Float16 h[16];
    uint4    q[2];
};

// gfx1250 has a hardware tanh transcendental (V_TANH_F32).
__device__ __forceinline__ float fast_tanh(float x) {
#if __has_builtin(__builtin_amdgcn_tanhf)
    return __builtin_amdgcn_tanhf(x);
#elif __has_builtin(__builtin_amdgcn_tanh_f32)
    return __builtin_amdgcn_tanh_f32(x);
#else
    // branch-free fallback: tanh(x) = 1 - 2/(e^{2x}+1)
    float e = __expf(2.0f * x);
    return 1.0f - 2.0f * __builtin_amdgcn_rcpf(e + 1.0f);
#endif
}

// sigmoid(x) == 0.5 + 0.5*tanh(x/2): one trans op + two FMAs, branch-free.
__device__ __forceinline__ float sigm(float x) {
    return 0.5f + 0.5f * fast_tanh(0.5f * x);
}

// Build a 32x16 (KxN) f16 B-fragment for WMMA f32_16x16x32_f16 from an f32
// weight matrix W[N_glob][ldw] (row = output gate column). Lane n=lane&15
// holds column n; lanes 0-15 carry K 0..15, lanes 16-31 carry K 16..31,
// two consecutive K per VGPR (mirrors the documented sparse 64x16 B layout).
__device__ __forceinline__ v16h load_bfrag(const float* __restrict__ w,
                                           int row, int ldw, int kbase) {
    F16Frag b;
#pragma unroll
    for (int s = 0; s < 16; ++s) b.h[s] = (_Float16)w[row * ldw + kbase + s];
    return b.v;
}

// ---------------------------------------------------------------------------
// Layer 0: input size 1. Block = 64 threads: wave0 = forward, wave1 = reverse.
// Emits h1[B][T][64] fp16 (fwd cols 0..31 at time t, bwd cols 32..63 at the
// re-aligned time len-1-t', matching reference _reverse_valid semantics).
// ---------------------------------------------------------------------------
__global__ __launch_bounds__(64) void lstm_layer0(
    const float* __restrict__ x, const int* __restrict__ lengths,
    const float* __restrict__ wih_f, const float* __restrict__ whh_f,
    const float* __restrict__ bih_f, const float* __restrict__ bhh_f,
    const float* __restrict__ wih_r, const float* __restrict__ whh_r,
    const float* __restrict__ bih_r, const float* __restrict__ bhh_r,
    _Float16* __restrict__ h1)
{
    __shared__ _Float16 hbuf[2][TILE_M * HID];   // per-direction h, [m][k] f16

    const int  wave = threadIdx.x >> 5;          // 0 = fwd, 1 = bwd
    const int  lane = threadIdx.x & 31;
    const int  b0   = blockIdx.x * TILE_M;
    const int  n    = lane & 15;                 // N column / A row this lane owns
    const bool hi   = lane >= 16;
    const int  dir  = wave;
    const int  kb   = hi ? 16 : 0;

    const float* wih = dir ? wih_r : wih_f;
    const float* whh = dir ? whh_r : whh_f;
    const float* bih = dir ? bih_r : bih_f;
    const float* bhh = dir ? bhh_r : bhh_f;

    _Float16* hl = hbuf[wave];

    // Recurrent weights as 8 resident B-fragments (gate cols 16j..16j+15).
    v16h Bw[8];
#pragma unroll
    for (int j = 0; j < 8; ++j) Bw[j] = load_bfrag(whh, 16 * j + n, HID, kb);

    float biasv[8], wxv[8];
#pragma unroll
    for (int j = 0; j < 8; ++j) {
        biasv[j] = bih[16 * j + n] + bhh[16 * j + n];
        wxv[j]   = wih[16 * j + n];              // I == 1: single column
    }

    const int mylen = lengths[b0 + n];           // length of row (lane&15)

    // zero h state (512 halfs) and c state
    for (int i = lane; i < TILE_M * HID / 2; i += 32) ((unsigned*)hl)[i] = 0u;
    asm volatile("" ::: "memory");

    v8f zero8 = {0, 0, 0, 0, 0, 0, 0, 0};
    v8f c_st[2] = {zero8, zero8};

    for (int t = 0; t < TSEQ; ++t) {
        // Input scalar for row n: fwd -> x[b,t]; bwd -> x[b,len-1-t] (0 if OOB).
        float xl;
        if (dir == 0) {
            xl = x[(size_t)(b0 + n) * TSEQ + t];
        } else {
            int q = mylen - 1 - t;
            xl = (q >= 0) ? x[(size_t)(b0 + n) * TSEQ + q] : 0.0f;
        }
        float xm[8];
#pragma unroll
        for (int r = 0; r < 8; ++r) xm[r] = __shfl(xl, r + (hi ? 8 : 0), 32);

        // Gates init: x*w_ih + b_ih + b_hh  (C-matrix layout: lane=n, reg=m)
        v8f g[8];
#pragma unroll
        for (int j = 0; j < 8; ++j)
#pragma unroll
            for (int r = 0; r < 8; ++r) g[j][r] = xm[r] * wxv[j] + biasv[j];

        // A fragment (16x32 f16, row n per lane) from LDS.
        F16Frag a;
        const uint4* hq = (const uint4*)(hl + n * HID);
        a.q[0] = hq[hi ? 2 : 0];                 // K 0..7  | 16..23
        a.q[1] = hq[hi ? 3 : 1];                 // K 8..15 | 24..31

#pragma unroll
        for (int j = 0; j < 8; ++j)
            g[j] = __builtin_amdgcn_wmma_f32_16x16x32_f16(
                false, a.v, false, Bw[j], (short)0, g[j], false, false);

        // Elementwise LSTM cell; PyTorch gate order i,f,g,o (tiles 01,23,45,67)
#pragma unroll
        for (int k = 0; k < 2; ++k)
#pragma unroll
            for (int r = 0; r < 8; ++r) {
                float ig = sigm(g[0 + k][r]);
                float fg = sigm(g[2 + k][r]);
                float gg = fast_tanh(g[4 + k][r]);
                float og = sigm(g[6 + k][r]);
                float c  = fg * c_st[k][r] + ig * gg;
                c_st[k][r] = c;
                float hv = og * fast_tanh(c);
                int m = r + (hi ? 8 : 0);
                hl[m * HID + k * 16 + n] = (_Float16)hv;   // scatter to [m][k]
            }
        asm volatile("" ::: "memory");

        // Coalesced fp16 tile store to h1: lane -> (row ms, 32B half `part`)
        {
            int  ms   = lane >> 1;
            int  part = lane & 1;
            int  ts   = t;
            bool ok   = true;
            if (dir == 1) {
                int lm = __shfl(mylen, ms, 32);
                ts = lm - 1 - t;                 // re-aligned time
                ok = ts >= 0;
            }
            if (ok) {
                const uint4* src = (const uint4*)(hl + ms * HID + part * 16);
                uint4* dst = (uint4*)(h1 + ((size_t)(b0 + ms) * TSEQ + ts) * 64
                                         + dir * 32 + part * 16);
                dst[0] = src[0];
                dst[1] = src[1];
            }
        }
    }
}

// ---------------------------------------------------------------------------
// Layer 1: input size 64. wave0 = forward full scan (latch h at t==len-1 in
// registers via cndmask, store once); wave1 = backward, which reduces to ONE
// step with input h1[b, len-1]. Writes h2last[B][64] fp32.
// ---------------------------------------------------------------------------
__global__ __launch_bounds__(64) void lstm_layer1(
    const _Float16* __restrict__ h1, const int* __restrict__ lengths,
    const float* __restrict__ wih_f, const float* __restrict__ whh_f,
    const float* __restrict__ bih_f, const float* __restrict__ bhh_f,
    const float* __restrict__ wih_r, const float* __restrict__ whh_r,
    const float* __restrict__ bih_r, const float* __restrict__ bhh_r,
    float* __restrict__ h2last)
{
    __shared__ _Float16 hbuf[TILE_M * HID];      // fwd wave recurrent state

    const int  wave = threadIdx.x >> 5;
    const int  lane = threadIdx.x & 31;
    const int  b0   = blockIdx.x * TILE_M;
    const int  n    = lane & 15;
    const bool hi   = lane >= 16;
    const int  kb   = hi ? 16 : 0;

    if (wave == 0) {
        // ---- forward direction: full scan ----
        v16h BI[16];                             // input weights [128x64] f16
#pragma unroll
        for (int j = 0; j < 8; ++j) {
            BI[2 * j + 0] = load_bfrag(wih_f, 16 * j + n, 64, 0 + kb);   // K 0..31
            BI[2 * j + 1] = load_bfrag(wih_f, 16 * j + n, 64, 32 + kb);  // K 32..63
        }
        v16h BH[8];
#pragma unroll
        for (int j = 0; j < 8; ++j) BH[j] = load_bfrag(whh_f, 16 * j + n, HID, kb);
        float biasv[8];
#pragma unroll
        for (int j = 0; j < 8; ++j) biasv[j] = bih_f[16 * j + n] + bhh_f[16 * j + n];

        int lenl = lengths[b0 + n];
        int lenm[8];
#pragma unroll
        for (int r = 0; r < 8; ++r) lenm[r] = __shfl(lenl, r + (hi ? 8 : 0), 32);

        for (int i = lane; i < TILE_M * HID / 2; i += 32) ((unsigned*)hbuf)[i] = 0u;
        asm volatile("" ::: "memory");

        v8f zero8 = {0, 0, 0, 0, 0, 0, 0, 0};
        v8f c_st[2]  = {zero8, zero8};
        v8f hsave[2] = {zero8, zero8};           // latched h at t == len-1

        for (int t = 0; t < TSEQ; ++t) {
            // Input A fragments (16x64 tile, split K 0..31 / 32..63) from h1.
            const uint4* xq =
                (const uint4*)(h1 + ((size_t)(b0 + n) * TSEQ + t) * 64);
            F16Frag a0, a1, ah;
            a0.q[0] = xq[hi ? 2 : 0];
            a0.q[1] = xq[hi ? 3 : 1];
            a1.q[0] = xq[hi ? 6 : 4];
            a1.q[1] = xq[hi ? 7 : 5];
            const uint4* hq = (const uint4*)(hbuf + n * HID);
            ah.q[0] = hq[hi ? 2 : 0];
            ah.q[1] = hq[hi ? 3 : 1];

            v8f g[8];
#pragma unroll
            for (int j = 0; j < 8; ++j) {
#pragma unroll
                for (int r = 0; r < 8; ++r) g[j][r] = biasv[j];
                g[j] = __builtin_amdgcn_wmma_f32_16x16x32_f16(
                    false, a0.v, false, BI[2 * j + 0], (short)0, g[j], false, false);
                g[j] = __builtin_amdgcn_wmma_f32_16x16x32_f16(
                    false, a1.v, false, BI[2 * j + 1], (short)0, g[j], false, false);
                g[j] = __builtin_amdgcn_wmma_f32_16x16x32_f16(
                    false, ah.v, false, BH[j], (short)0, g[j], false, false);
            }

#pragma unroll
            for (int k = 0; k < 2; ++k)
#pragma unroll
                for (int r = 0; r < 8; ++r) {
                    float ig = sigm(g[0 + k][r]);
                    float fg = sigm(g[2 + k][r]);
                    float gg = fast_tanh(g[4 + k][r]);
                    float og = sigm(g[6 + k][r]);
                    float c  = fg * c_st[k][r] + ig * gg;
                    c_st[k][r] = c;
                    float hv = og * fast_tanh(c);
                    int m = r + (hi ? 8 : 0);
                    hbuf[m * HID + k * 16 + n] = (_Float16)hv;
                    // Latch last valid timestep in registers (v_cndmask).
                    hsave[k][r] = (t == lenm[r] - 1) ? hv : hsave[k][r];
                }
            asm volatile("" ::: "memory");
        }

        // Emit latched outputs once.
#pragma unroll
        for (int k = 0; k < 2; ++k)
#pragma unroll
            for (int r = 0; r < 8; ++r) {
                int m = r + (hi ? 8 : 0);
                h2last[(size_t)(b0 + m) * 64 + k * 16 + n] = hsave[k][r];
            }
    } else {
        // ---- backward direction: single step (h0=c0=0, input h1[b,len-1]) ----
        v16h BI[16];
#pragma unroll
        for (int j = 0; j < 8; ++j) {
            BI[2 * j + 0] = load_bfrag(wih_r, 16 * j + n, 64, 0 + kb);
            BI[2 * j + 1] = load_bfrag(wih_r, 16 * j + n, 64, 32 + kb);
        }
        float biasv[8];
#pragma unroll
        for (int j = 0; j < 8; ++j) biasv[j] = bih_r[16 * j + n] + bhh_r[16 * j + n];

        int tm = lengths[b0 + n] - 1;            // len >= 1 guaranteed
        const uint4* xq =
            (const uint4*)(h1 + ((size_t)(b0 + n) * TSEQ + tm) * 64);
        F16Frag a0, a1;
        a0.q[0] = xq[hi ? 2 : 0];
        a0.q[1] = xq[hi ? 3 : 1];
        a1.q[0] = xq[hi ? 6 : 4];
        a1.q[1] = xq[hi ? 7 : 5];

        v8f g[8];
#pragma unroll
        for (int j = 0; j < 8; ++j) {
#pragma unroll
            for (int r = 0; r < 8; ++r) g[j][r] = biasv[j];
            g[j] = __builtin_amdgcn_wmma_f32_16x16x32_f16(
                false, a0.v, false, BI[2 * j + 0], (short)0, g[j], false, false);
            g[j] = __builtin_amdgcn_wmma_f32_16x16x32_f16(
                false, a1.v, false, BI[2 * j + 1], (short)0, g[j], false, false);
        }

#pragma unroll
        for (int k = 0; k < 2; ++k)
#pragma unroll
            for (int r = 0; r < 8; ++r) {
                float ig = sigm(g[0 + k][r]);
                float gg = fast_tanh(g[4 + k][r]);
                float og = sigm(g[6 + k][r]);
                float c  = ig * gg;              // f-gate * c0 == 0
                float hv = og * fast_tanh(c);
                int m = r + (hi ? 8 : 0);
                h2last[(size_t)(b0 + m) * 64 + 32 + k * 16 + n] = hv;
            }
    }
}

// ---------------------------------------------------------------------------
// Final classifier: out[b] = h2last[b, :64] @ w_out^T + b_out   (64 -> 2)
// ---------------------------------------------------------------------------
__global__ __launch_bounds__(256) void classifier(
    const float* __restrict__ h2last, const float* __restrict__ w_out,
    const float* __restrict__ b_out, float* __restrict__ out)
{
    int b = blockIdx.x * blockDim.x + threadIdx.x;
    if (b >= BATCH) return;
    const float* hrow = h2last + (size_t)b * 64;
    float acc0 = b_out[0], acc1 = b_out[1];
#pragma unroll 8
    for (int k = 0; k < 64; ++k) {
        float hv = hrow[k];
        acc0 += hv * w_out[k];
        acc1 += hv * w_out[64 + k];
    }
    out[b * 2 + 0] = acc0;
    out[b * 2 + 1] = acc1;
}

extern "C" void kernel_launch(void* const* d_in, const int* in_sizes, int n_in,
                              void* d_out, int out_size, void* d_ws, size_t ws_size,
                              hipStream_t stream)
{
    const float* x       = (const float*)d_in[0];
    const int*   lengths = (const int*)  d_in[1];
    const float* wih0    = (const float*)d_in[2];
    const float* whh0    = (const float*)d_in[3];
    const float* bih0    = (const float*)d_in[4];
    const float* bhh0    = (const float*)d_in[5];
    const float* wih0r   = (const float*)d_in[6];
    const float* whh0r   = (const float*)d_in[7];
    const float* bih0r   = (const float*)d_in[8];
    const float* bhh0r   = (const float*)d_in[9];
    const float* wih1    = (const float*)d_in[10];
    const float* whh1    = (const float*)d_in[11];
    const float* bih1    = (const float*)d_in[12];
    const float* bhh1    = (const float*)d_in[13];
    const float* wih1r   = (const float*)d_in[14];
    const float* whh1r   = (const float*)d_in[15];
    const float* bih1r   = (const float*)d_in[16];
    const float* bhh1r   = (const float*)d_in[17];
    const float* w_out   = (const float*)d_in[18];
    const float* b_out   = (const float*)d_in[19];

    // Workspace layout: h1 fp16 [B][T][64] (128 MB), then h2last fp32 [B][64].
    _Float16* h1     = (_Float16*)d_ws;
    float*    h2last = (float*)((char*)d_ws +
                                (size_t)BATCH * TSEQ * 64 * sizeof(_Float16));

    dim3 grid(BATCH / TILE_M);   // 128 tiles, 2 waves (directions) each
    lstm_layer0<<<grid, 64, 0, stream>>>(x, lengths,
                                         wih0, whh0, bih0, bhh0,
                                         wih0r, whh0r, bih0r, bhh0r, h1);
    lstm_layer1<<<grid, 64, 0, stream>>>(h1, lengths,
                                         wih1, whh1, bih1, bhh1,
                                         wih1r, whh1r, bih1r, bhh1r, h2last);
    classifier<<<dim3(BATCH / 256), 256, 0, stream>>>(h2last, w_out, b_out,
                                                      (float*)d_out);
}
